// MultiHeadAttentionRel_85839216377912
// MI455X (gfx1250) — matrix operand order repaired and checked by reading, not compiled, not run on hardware
//
#include <hip/hip_runtime.h>
#include <hip/hip_bf16.h>

// MI455X / gfx1250 fused relative-position multi-head attention.
// Pipeline: qkv_kernel (f16 WMMA GEMM) -> attn_kernel (flash-style, 2-pass
// softmax, rel-K/rel-V via small WMMA GEMMs, TDM-staged K/V tiles in LDS)
// -> outproj_kernel (WMMA GEMM).  Matmuls use v_wmma_f32_16x16x32_f16.

typedef _Float16 h16;
typedef __attribute__((ext_vector_type(8)))  _Float16 v8h;
typedef __attribute__((ext_vector_type(16))) _Float16 v16h;
typedef __attribute__((ext_vector_type(8)))  float    v8f;

typedef unsigned int u32;
typedef __attribute__((ext_vector_type(4))) unsigned int u32x4;
typedef __attribute__((ext_vector_type(8))) int          i32x8;
typedef __attribute__((ext_vector_type(4))) int          i32x4;

#define Bsz 8
#define Csz 512
#define Tsz 1024
#define Hsz 8
#define Dsz 64
#define RELN 129
#define RPAD 160   // 129 padded to multiple of 32 for the rel-V GEMM

static __device__ inline v8f wmma16(v16h a, v16h b, v8f c) {
  return __builtin_amdgcn_wmma_f32_16x16x32_f16(false, a, false, b, (short)0, c,
                                                false, false);
}

// ---- TDM: async 2-D tile load global->LDS (CDNA5 ISA ch. 7/8) -------------
// D# group0: count=1 (load, user), lds_addr[63:32], global_addr[120:64],
//            type=2 in [127:126].
// D# group1: data_size=1 (2 bytes) in [17:16]; tensor_dim0 [79:48];
//            tensor_dim1 [111:80]; tile_dim0 [127:112]; tile_dim1 [143:128];
//            tensor_dim0_stride [207:160].
static __device__ inline void tdm_load_2d(u32 lds_addr, const void* gptr,
                                          u32 tile0, u32 tile1, u32 stride0) {
  unsigned long long ga = (unsigned long long)(uintptr_t)gptr;
  const u32 tdim0 = 1u << 20, tdim1 = 1u << 20;  // tiles are interior: no OOB
  u32x4 g0;
  g0[0] = 1u;                                    // count=1, load, user mode
  g0[1] = lds_addr;
  g0[2] = (u32)ga;
  g0[3] = ((u32)(ga >> 32) & 0x01FFFFFFu) | (2u << 30);  // addr[56:32], type=2
  i32x8 g1;
  g1[0] = (int)(1u << 16);                                   // data_size = 2B
  g1[1] = (int)((tdim0 & 0xFFFFu) << 16);
  g1[2] = (int)(((tdim0 >> 16) & 0xFFFFu) | ((tdim1 & 0xFFFFu) << 16));
  g1[3] = (int)(((tdim1 >> 16) & 0xFFFFu) | ((tile0 & 0xFFFFu) << 16));
  g1[4] = (int)(tile1 & 0xFFFFu);
  g1[5] = (int)stride0;
  g1[6] = 0;
  g1[7] = 0;
  i32x4 z4;
  z4[0] = z4[1] = z4[2] = z4[3] = 0;
#if defined(__clang_major__) && (__clang_major__ >= 23)
  i32x8 z8;
  z8[0] = z8[1] = z8[2] = z8[3] = z8[4] = z8[5] = z8[6] = z8[7] = 0;
  __builtin_amdgcn_tensor_load_to_lds(g0, g1, z4, z4, z8, 0);
#else
  __builtin_amdgcn_tensor_load_to_lds(g0, g1, z4, z4, 0);
#endif
}

// Flat LDS pointers keep the byte offset in addr[31:0].
static __device__ inline u32 lds_off(const void* p) {
  return (u32)(uintptr_t)p;
}

// ---- Fragment loaders (wave32; A/B use contiguous k order, not the ISA table order) ----------------
// A (16x32 f16): lane holds row (lane&15). halves sequential:
//   K = k0 + (hi?16:0) + h.  hi = lane>=16.
// B (32x16 f16): lane holds col (lane&15). halves sequential:
//   K = k0 + (hi?16:0) + h.
// C/D (16x16 f32): VGPR e: row = e + (hi?8:0), col = lane&15.

static __device__ inline v16h load_a_h16(const h16* src, int ld, int row,
                                         int k0, int lane) {
  const h16* p = src + row * ld + k0 + ((lane & 16) ? 16 : 0);
  v8h lo = *(const v8h*)(p);
  v8h hi = *(const v8h*)(p + 8);
  v16h a;
#pragma unroll
  for (int i = 0; i < 8; ++i) { a[i] = lo[i]; a[i + 8] = hi[i]; }
  return a;
}

static __device__ inline v16h load_a_f32(const float* src, int ld, int row,
                                         int k0, int lane) {
  const float* p = src + row * ld + k0 + ((lane & 16) ? 16 : 0);
  v16h a;
#pragma unroll
  for (int i = 0; i < 8; ++i) { a[i] = (h16)p[i]; a[i + 8] = (h16)p[8 + i]; }
  return a;
}

// A-frag: element = erv[k*64 + row] (emb_rel_v^T), guard k < 129
static __device__ inline v16h load_a_ervT(const float* erv, int row, int k0,
                                          int lane) {
  int kb = k0 + ((lane & 16) ? 16 : 0);
  v16h a;
#pragma unroll
  for (int i = 0; i < 8; ++i) {
    int r0 = kb + i, r1 = kb + 8 + i;
    a[i]     = (r0 < RELN) ? (h16)erv[r0 * Dsz + row] : (h16)0.0f;
    a[i + 8] = (r1 < RELN) ? (h16)erv[r1 * Dsz + row] : (h16)0.0f;
  }
  return a;
}

// B-frag from f16 "N-major" src[n*ld + k]  (contiguous halves per lane)
static __device__ inline v16h load_b_h16_nmajor(const h16* src, int ld, int n,
                                                int k0, int lane) {
  const h16* p = src + n * ld + k0 + ((lane & 16) ? 16 : 0);
  v8h lo = *(const v8h*)(p);
  v8h hi = *(const v8h*)(p + 8);
  v16h b;
#pragma unroll
  for (int i = 0; i < 8; ++i) { b[i] = lo[i]; b[i + 8] = hi[i]; }
  return b;
}

static __device__ inline v16h load_b_f32_nmajor(const float* src, int ld, int n,
                                                int k0, int lane) {
  const float* p = src + n * ld + k0 + ((lane & 16) ? 16 : 0);
  v16h b;
#pragma unroll
  for (int i = 0; i < 16; ++i) b[i] = (h16)p[i];
  return b;
}

static __device__ inline v16h load_b_f32_kmajor(const float* src, int ld,
                                                int k0, int n, int lane) {
  int kb = k0 + ((lane & 16) ? 16 : 0);
  v16h b;
#pragma unroll
  for (int i = 0; i < 16; ++i) b[i] = (h16)src[(kb + i) * ld + n];
  return b;
}

static __device__ inline v16h load_b_h16_kmajor(const h16* src, int ld, int k0,
                                                int n, int lane) {
  int kb = k0 + ((lane & 16) ? 16 : 0);
  v16h b;
#pragma unroll
  for (int i = 0; i < 16; ++i) b[i] = src[(kb + i) * ld + n];
  return b;
}

// B-frag: element = erk[n*64 + k] (n = rel index, contiguous in d), guard n<129
static __device__ inline v16h load_b_erk(const float* erk, int n0, int k0,
                                         int lane) {
  int r = n0 + (lane & 15);
  int kb = k0 + ((lane & 16) ? 16 : 0);
  v16h b;
  if (r < RELN) {
    const float* p = erk + r * Dsz + kb;
#pragma unroll
    for (int i = 0; i < 16; ++i) b[i] = (h16)p[i];
  } else {
#pragma unroll
    for (int i = 0; i < 16; ++i) b[i] = (h16)0.0f;
  }
  return b;
}

// ---- Kernel 1: QKV projection --------------------------------------------
__global__ __launch_bounds__(128) void qkv_kernel(
    const float* __restrict__ x,
    const float* __restrict__ Wq, const float* __restrict__ bq,
    const float* __restrict__ Wk, const float* __restrict__ bk,
    const float* __restrict__ Wv, const float* __restrict__ bv,
    h16* __restrict__ q, h16* __restrict__ k, h16* __restrict__ vT) {
  const int lane = threadIdx.x & 31;
  const int wtile = blockIdx.x * 4 + (threadIdx.x >> 5);  // 3*8*2048 waves
  const int m = wtile / (Bsz * 2048);
  const int rem = wtile % (Bsz * 2048);
  const int b = rem / 2048;
  const int t2 = rem % 2048;
  const int o0 = (t2 >> 6) * 16;
  const int t0 = (t2 & 63) * 16;

  const float* W    = (m == 0) ? Wq : ((m == 1) ? Wk : Wv);
  const float* bias = (m == 0) ? bq : ((m == 1) ? bk : bv);
  const float* xb = x + b * Csz * Tsz;
  const int row = o0 + (lane & 15);
  const int col = t0 + (lane & 15);

  v8f acc = {};
  for (int c0 = 0; c0 < Csz; c0 += 32) {
    if (c0 + 32 < Csz)
      __builtin_prefetch(xb + (c0 + 32) * Tsz + col, 0, 0);
    v16h a  = load_a_f32(W, Csz, row, c0, lane);
    v16h bb = load_b_f32_kmajor(xb, Tsz, c0, col, lane);
    acc = wmma16(a, bb, acc);
  }

  const int off = (lane & 16) ? 8 : 0;
#pragma unroll
  for (int e = 0; e < 8; ++e) {
    int o = o0 + e + off;
    h16 hv = (h16)(acc[e] + bias[o]);
    if (m == 2) {
      vT[(b * Csz + o) * Tsz + col] = hv;
    } else {
      h16* dst = (m == 0) ? q : k;
      dst[((b * Hsz + (o >> 6)) * Tsz + col) * Dsz + (o & 63)] = hv;
    }
  }
}

// ---- Kernel 2: fused relative attention ----------------------------------
// One wave per 16-row query tile; 2 waves per workgroup; both waves share the
// same (b,h), so K/V tiles are staged once per WG in LDS via the Tensor Data
// Mover (double-buffered, TENSORcnt-synchronized).
__global__ __launch_bounds__(64) void attn_kernel(
    const h16* __restrict__ q, const h16* __restrict__ k,
    const h16* __restrict__ vT, const float* __restrict__ erk,
    const float* __restrict__ erv, const float* __restrict__ xmask,
    h16* __restrict__ yc) {
  __shared__ __align__(16) float r_lds[2][16 * RPAD];  // R[i,r] = q . rel_k
  __shared__ __align__(16) float w_lds[2][16 * RPAD];  // W_rel[i,r]
  __shared__ __align__(16) h16   p_lds[2][16 * 32];    // P^T staging
  __shared__ __align__(16) float inv_l[2][16];         // 1/rowsum
  __shared__ __align__(16) h16   kstage[2][32 * 64];   // K tile (32 t x 64 d)
  __shared__ __align__(16) h16   vstage[2][64 * 32];   // V^T tile (64 d x 32 t)

  const int lane = threadIdx.x & 31;
  const int wave = threadIdx.x >> 5;
  float* R   = r_lds[wave];
  float* Wr  = w_lds[wave];
  h16*   P   = p_lds[wave];
  float* INV = inv_l[wave];

  const int tile = blockIdx.x * 2 + wave;  // 0..4095
  const int bh = tile >> 6;                // uniform across the WG
  const int i0 = (tile & 63) * 16;
  const int b = bh >> 3;

  const h16* qp = q  + bh * Tsz * Dsz;
  const h16* kp = k  + bh * Tsz * Dsz;
  const h16* vp = vT + bh * Dsz * Tsz;  // rows h*64+d of (B,C,T)
  h16*       yp = yc + bh * Dsz * Tsz;
  const float* xm = xmask + b * Tsz;

  const int off = (lane & 16) ? 8 : 0;
  const int nlo = lane & 15;

  for (int idx = lane; idx < 16 * RPAD; idx += 32) { R[idx] = 0.f; Wr[idx] = 0.f; }

  const int qrow = i0 + nlo;
  v16h qa0 = load_a_h16(qp, Dsz, qrow, 0, lane);
  v16h qa1 = load_a_h16(qp, Dsz, qrow, 32, lane);

  // R[i,r] = sum_d q[i,d] * erk[r,d]   (9 tiles of 16 rel columns)
  for (int rt = 0; rt < 9; ++rt) {
    int r0 = rt * 16;
    v8f acc = {};
    acc = wmma16(qa0, load_b_erk(erk, r0, 0, lane), acc);
    acc = wmma16(qa1, load_b_erk(erk, r0, 32, lane), acc);
#pragma unroll
    for (int e = 0; e < 8; ++e) R[(e + off) * RPAD + r0 + nlo] = acc[e];
  }
  __syncthreads();

  // ---- pass 1: row maxima (K tiles TDM-staged, double-buffered) -----------
  float vmax[8];
#pragma unroll
  for (int e = 0; e < 8; ++e) vmax[e] = -3.0e38f;

  if (wave == 0)
    tdm_load_2d(lds_off(&kstage[0][0]), kp, 64u, 32u, 64u);

  for (int jp = 0; jp < 32; ++jp) {
    int j0 = jp * 32;
    if (wave == 0) {
      if (jp + 1 < 32) {
        tdm_load_2d(lds_off(&kstage[(jp + 1) & 1][0]), kp + (j0 + 32) * Dsz,
                    64u, 32u, 64u);
        __builtin_amdgcn_s_wait_tensorcnt(1);
      } else {
        __builtin_amdgcn_s_wait_tensorcnt(0);
      }
    }
    __syncthreads();
    const h16* kst = kstage[jp & 1];
#pragma unroll
    for (int sidx = 0; sidx < 2; ++sidx) {
      int jj0 = j0 + sidx * 16;
      v8f acc = {};
      acc = wmma16(qa0, load_b_h16_nmajor(kst, Dsz, sidx * 16 + nlo, 0, lane), acc);
      acc = wmma16(qa1, load_b_h16_nmajor(kst, Dsz, sidx * 16 + nlo, 32, lane), acc);
      int jcol = jj0 + nlo;
      bool valid = xm[jcol] > 0.f;
#pragma unroll
      for (int e = 0; e < 8; ++e) {
        int il = e + off;
        int rel = jcol - (i0 + il);
        rel = rel < -64 ? -64 : (rel > 64 ? 64 : rel);
        float s = acc[e] * 0.125f + R[il * RPAD + rel + 64];
        if (!valid) s = -1.0e9f;
        vmax[e] = fmaxf(vmax[e], s);
      }
    }
    __syncthreads();
  }
#pragma unroll
  for (int e = 0; e < 8; ++e)
#pragma unroll
    for (int msk = 1; msk < 16; msk <<= 1)
      vmax[e] = fmaxf(vmax[e], __shfl_xor(vmax[e], msk, 32));

  // ---- pass 2: probabilities, P.V, W_rel (K and V TDM-staged) -------------
  float lsum[8];
#pragma unroll
  for (int e = 0; e < 8; ++e) lsum[e] = 0.f;
  v8f yacc[4];
#pragma unroll
  for (int dc = 0; dc < 4; ++dc) { v8f z = {}; yacc[dc] = z; }

  if (wave == 0) {
    tdm_load_2d(lds_off(&kstage[0][0]), kp, 64u, 32u, 64u);
    tdm_load_2d(lds_off(&vstage[0][0]), vp, 32u, 64u, (u32)Tsz);
  }

  for (int jp = 0; jp < 32; ++jp) {
    int j0 = jp * 32;
    if (wave == 0) {
      if (jp + 1 < 32) {
        tdm_load_2d(lds_off(&kstage[(jp + 1) & 1][0]), kp + (j0 + 32) * Dsz,
                    64u, 32u, 64u);
        tdm_load_2d(lds_off(&vstage[(jp + 1) & 1][0]), vp + (j0 + 32),
                    32u, 64u, (u32)Tsz);
        __builtin_amdgcn_s_wait_tensorcnt(2);
      } else {
        __builtin_amdgcn_s_wait_tensorcnt(0);
      }
    }
    __syncthreads();
    const h16* kst = kstage[jp & 1];
    const h16* vst = vstage[jp & 1];
#pragma unroll
    for (int sidx = 0; sidx < 2; ++sidx) {
      int jj0 = j0 + sidx * 16;
      v8f acc = {};
      acc = wmma16(qa0, load_b_h16_nmajor(kst, Dsz, sidx * 16 + nlo, 0, lane), acc);
      acc = wmma16(qa1, load_b_h16_nmajor(kst, Dsz, sidx * 16 + nlo, 32, lane), acc);
      int jcol = jj0 + nlo;
      bool valid = xm[jcol] > 0.f;
#pragma unroll
      for (int e = 0; e < 8; ++e) {
        int il = e + off;
        int rel = jcol - (i0 + il);
        rel = rel < -64 ? -64 : (rel > 64 ? 64 : rel);
        float s = acc[e] * 0.125f + R[il * RPAD + rel + 64];
        if (!valid) s = -1.0e9f;
        float p = __expf(s - vmax[e]);
        lsum[e] += p;
        atomicAdd(&Wr[il * RPAD + rel + 64], p);
        P[il * 32 + sidx * 16 + nlo] = (h16)p;
      }
    }
    __syncthreads();
    // y^T += V^T (16d x 32j) * P^T (32j x 16i)
    v16h pb = load_b_h16_nmajor(P, 32, nlo, 0, lane);
#pragma unroll
    for (int dc = 0; dc < 4; ++dc) {
      v16h va = load_a_h16(vst, 32, dc * 16 + nlo, 0, lane);
      yacc[dc] = wmma16(va, pb, yacc[dc]);
    }
    __syncthreads();
  }

  // row sums -> 1/l
#pragma unroll
  for (int e = 0; e < 8; ++e)
#pragma unroll
    for (int msk = 1; msk < 16; msk <<= 1)
      lsum[e] += __shfl_xor(lsum[e], msk, 32);
  if (nlo == 0) {
#pragma unroll
    for (int e = 0; e < 8; ++e) INV[e + off] = 1.0f / lsum[e];
  }
  __syncthreads();
  float invc = INV[nlo];
#pragma unroll
  for (int dc = 0; dc < 4; ++dc)
#pragma unroll
    for (int e = 0; e < 8; ++e) yacc[dc][e] *= invc;

  // normalize W_rel rows by 1/l
  for (int idx = lane; idx < 16 * RPAD; idx += 32) Wr[idx] *= INV[idx / RPAD];
  __syncthreads();

  // y^T += erv^T (16d x 32r) * W_rel^T (32r x 16i)
  for (int rc = 0; rc < 5; ++rc) {
    int r0 = rc * 32;
    v16h bW = load_b_f32_nmajor(Wr, RPAD, nlo, r0, lane);
#pragma unroll
    for (int dc = 0; dc < 4; ++dc) {
      v16h aE = load_a_ervT(erv, dc * 16 + nlo, r0, lane);
      yacc[dc] = wmma16(aE, bW, yacc[dc]);
    }
  }

  // store y^T directly into (B, C, T) f16 buffer for the output GEMM
#pragma unroll
  for (int dc = 0; dc < 4; ++dc)
#pragma unroll
    for (int e = 0; e < 8; ++e) {
      int d = dc * 16 + e + off;
      yp[d * Tsz + i0 + nlo] = (h16)yacc[dc][e];
    }
}

// ---- Kernel 3: output projection + mask ----------------------------------
__global__ __launch_bounds__(128) void outproj_kernel(
    const h16* __restrict__ yc, const float* __restrict__ Wo,
    const float* __restrict__ bo, const float* __restrict__ xmask,
    float* __restrict__ out) {
  const int lane = threadIdx.x & 31;
  const int wtile = blockIdx.x * 4 + (threadIdx.x >> 5);  // 8*2048 waves
  const int b = wtile / 2048;
  const int t2 = wtile % 2048;
  const int o0 = (t2 >> 6) * 16;
  const int t0 = (t2 & 63) * 16;

  const h16* ycb = yc + b * Csz * Tsz;
  const int row = o0 + (lane & 15);
  const int col = t0 + (lane & 15);

  v8f acc = {};
  for (int c0 = 0; c0 < Csz; c0 += 32) {
    v16h a  = load_a_f32(Wo, Csz, row, c0, lane);
    v16h bb = load_b_h16_kmajor(ycb, Tsz, c0, col, lane);
    acc = wmma16(a, bb, acc);
  }

  const float xmv = xmask[b * Tsz + col];
  const int off = (lane & 16) ? 8 : 0;
#pragma unroll
  for (int e = 0; e < 8; ++e) {
    int o = o0 + e + off;
    out[(b * Csz + o) * Tsz + col] = (acc[e] + bo[o]) * xmv;
  }
}

extern "C" void kernel_launch(void* const* d_in, const int* in_sizes, int n_in,
                              void* d_out, int out_size, void* d_ws,
                              size_t ws_size, hipStream_t stream) {
  (void)in_sizes; (void)n_in; (void)out_size; (void)ws_size;
  const float* x     = (const float*)d_in[0];
  const float* xmask = (const float*)d_in[1];
  const float* Wq = (const float*)d_in[2];
  const float* bq = (const float*)d_in[3];
  const float* Wk = (const float*)d_in[4];
  const float* bk = (const float*)d_in[5];
  const float* Wv = (const float*)d_in[6];
  const float* bv = (const float*)d_in[7];
  const float* Wo = (const float*)d_in[8];
  const float* bo = (const float*)d_in[9];
  const float* erk = (const float*)d_in[10];
  const float* erv = (const float*)d_in[11];

  // workspace: 4 x (B*C*T) f16 buffers = 4 x 8 MiB = 32 MiB
  const size_t N = (size_t)Bsz * Csz * Tsz;
  h16* q  = (h16*)d_ws;
  h16* k  = q + N;
  h16* vT = k + N;
  h16* yc = vT + N;

  qkv_kernel<<<dim3(12288), dim3(128), 0, stream>>>(x, Wq, bq, Wk, bk, Wv, bv,
                                                    q, k, vT);
  attn_kernel<<<dim3(2048), dim3(64), 0, stream>>>(q, k, vT, erk, erv, xmask,
                                                   yc);
  outproj_kernel<<<dim3(4096), dim3(128), 0, stream>>>(yc, Wo, bo, xmask,
                                                       (float*)d_out);
}
